// HGTLinkPredictor_25005299597850
// MI455X (gfx1250) — matrix operand order repaired and checked by reading, not compiled, not run on hardware
//
#include <hip/hip_runtime.h>
#include <hip/hip_bf16.h>

typedef _Float16 h4f  __attribute__((ext_vector_type(4)));
typedef _Float16 h8f  __attribute__((ext_vector_type(8)));
typedef _Float16 v16h __attribute__((ext_vector_type(16)));
typedef float    v8f  __attribute__((ext_vector_type(8)));

#define T_NT  3
#define R_ET  4
#define L_LAY 2
#define HHID  256
#define HEADS 8
#define DHEAD 32
#define OUTD  128
#define FIND  128
#define BKT   64    // K-tile staged in LDS per round (double-buffered)

__device__ __forceinline__ float gelu_exact(float x) {
  return 0.5f * x * (1.0f + erff(x * 0.70710678118654752f));
}

// order-preserving float<->uint mapping for atomicMax on floats
__device__ __forceinline__ unsigned fmap(float f) {
  unsigned u = __float_as_uint(f);
  return (u & 0x80000000u) ? ~u : (u | 0x80000000u);
}
__device__ __forceinline__ float funmap(unsigned u) {
  unsigned b = (u & 0x80000000u) ? (u ^ 0x80000000u) : ~u;
  return __uint_as_float(b);
}

// generic (flat) pointer -> 32-bit LDS offset (low 32 bits per flat aperture rules)
__device__ __forceinline__ unsigned lds_off(const void* p) {
  return (unsigned)(size_t)p;
}

__device__ __forceinline__ void async_load_b128(unsigned lds_addr, const void* gaddr) {
  asm volatile("global_load_async_to_lds_b128 %0, %1, off"
               :: "v"(lds_addr), "v"(gaddr) : "memory");
}

// ---------------------------------------------------------------------------
// f32 -> f16 conversion (vectorized), optional exact-GELU prologue fused in.
// ---------------------------------------------------------------------------
__global__ void cvt_f16_kernel(const float* __restrict__ in,
                               _Float16* __restrict__ out, long n4, int act)
{
  long gid = (long)blockIdx.x * blockDim.x + threadIdx.x;
  if (gid >= n4) return;
  float4 v = ((const float4*)in)[gid];
  if (act) {
    v.x = gelu_exact(v.x); v.y = gelu_exact(v.y);
    v.z = gelu_exact(v.z); v.w = gelu_exact(v.w);
  }
  h4f o;
  o[0] = (_Float16)v.x; o[1] = (_Float16)v.y;
  o[2] = (_Float16)v.z; o[3] = (_Float16)v.w;
  ((h4f*)out)[gid] = o;
}

// ---------------------------------------------------------------------------
// Weight convert + transpose: BT[mat][n][k] = (f16) B[mat][k][n]
// ---------------------------------------------------------------------------
__global__ void wtrans_f16_kernel(const float* __restrict__ Bsrc,
                                  _Float16* __restrict__ BTd,
                                  int K, int Nout, long total)
{
  long gid = (long)blockIdx.x * blockDim.x + threadIdx.x;
  if (gid >= total) return;
  long kn  = (long)K * Nout;
  long mat = gid / kn;
  long rem = gid - mat * kn;
  int n = (int)(rem / K);
  int k = (int)(rem - (long)n * K);
  BTd[gid] = (_Float16)Bsrc[mat * kn + (long)k * Nout + n];
}

// ---------------------------------------------------------------------------
// Tiled WMMA GEMM: C[z] = act(A[z] * BT[z] + bias[z])
// A:  [Mpad,K] f16 row-major (pre-converted; tile overreads land in pad)
// BT: [Nout,K] f16 (pre-transposed weights, contiguous in K)
// C:  [M,Nout] f32.  Block tile 128x64; K staged 64 per round into
// double-buffered LDS via async global->LDS, with the next tile's fetch
// issued before waiting on the current one (s_wait_asynccnt 6 drains only
// the in-flight tile; async loads complete in order). 8 waves; wave w owns
// rows [16w,16w+16); fragments batched so 4 WMMAs issue back-to-back.
// ---------------------------------------------------------------------------
__global__ __launch_bounds__(256) void gemm_wmma_kernel(
    const _Float16* __restrict__ A, const _Float16* __restrict__ BT,
    const float* __restrict__ bias, float* __restrict__ C,
    int M, int K, int Nout,
    long sA, long sB, long sBias, long sC, int actC)
{
  __shared__ __align__(16) _Float16 As[2][128 * BKT];  // 2 x 16 KB
  __shared__ __align__(16) _Float16 Bs[2][64 * BKT];   // 2 x  8 KB

  const int tid  = threadIdx.x;
  const int wave = tid >> 5;
  const int lane = tid & 31;
  const int rowBase = blockIdx.x * 128;
  const int colBase = blockIdx.y * 64;

  A    += (long)blockIdx.z * sA;
  BT   += (long)blockIdx.z * sB;
  bias += (long)blockIdx.z * sBias;
  C    += (long)blockIdx.z * sC;

  const unsigned ldsA = lds_off(&As[0][0]);
  const unsigned ldsB = lds_off(&Bs[0][0]);

  v8f acc[4] = {};

  const int ar  = (wave << 4) + (lane & 15);   // A row within tile
  const int kbA = (lane >> 4) << 3;            // 0 or 8  (A interleaved halves)
  const int kbB = (lane >> 4) << 4;            // 0 or 16 (unused; B also loads with kbA)

  // Issue one 128x64(A) + 64x64(B) f16 tile: 6 async b128 per thread.
  auto issue_tile = [&](int k0, int buf) {
    unsigned la = ldsA + (unsigned)buf * (128 * BKT * 2);
#pragma unroll
    for (int i = 0; i < 4; ++i) {
      int chunk = tid + i * 256;               // 0..1023; 8 chunks per row
      async_load_b128(la + (unsigned)chunk * 16,
                      A + (long)(rowBase + (chunk >> 3)) * K + k0 + ((chunk & 7) << 3));
    }
    unsigned lb = ldsB + (unsigned)buf * (64 * BKT * 2);
#pragma unroll
    for (int i = 0; i < 2; ++i) {
      int chunk = tid + i * 256;               // 0..511
      async_load_b128(lb + (unsigned)chunk * 16,
                      BT + (long)(colBase + (chunk >> 3)) * K + k0 + ((chunk & 7) << 3));
    }
  };

  int cur = 0;
  issue_tile(0, 0);

  for (int k0 = 0; k0 < K; k0 += BKT) {
    const bool more = (k0 + BKT) < K;
    if (more) issue_tile(k0 + BKT, cur ^ 1);   // prefetch next tile (other buffer)
    if (more) asm volatile("s_wait_asynccnt 0x6" ::: "memory");  // drain current tile only
    else      asm volatile("s_wait_asynccnt 0x0" ::: "memory");
    __syncthreads();

    const _Float16* pA = As[cur];
    const _Float16* pB = Bs[cur];
#pragma unroll
    for (int kk = 0; kk < BKT; kk += 32) {
      // A fragment: elems 0..7 = K kk+kbA.., elems 8..15 = K kk+16+kbA..
      h8f alo = *(const h8f*)&pA[ar * BKT + kk + kbA];
      h8f ahi = *(const h8f*)&pA[ar * BKT + kk + 16 + kbA];
      v16h afrag = __builtin_shufflevector(alo, ahi,
          0, 1, 2, 3, 4, 5, 6, 7, 8, 9, 10, 11, 12, 13, 14, 15);
      // Batch all 4 B fragments, then 4 back-to-back WMMAs (one dscnt wait).
      v16h bf[4];
#pragma unroll
      for (int c4 = 0; c4 < 4; ++c4) {
        int bn = (c4 << 4) + (lane & 15);
        h8f blo = *(const h8f*)&pB[bn * BKT + kk + kbA];
        h8f bhi = *(const h8f*)&pB[bn * BKT + kk + 16 + kbA];
        bf[c4] = __builtin_shufflevector(blo, bhi,
            0, 1, 2, 3, 4, 5, 6, 7, 8, 9, 10, 11, 12, 13, 14, 15);
      }
#pragma unroll
      for (int c4 = 0; c4 < 4; ++c4) {
        acc[c4] = __builtin_amdgcn_wmma_f32_16x16x32_f16(
            false, afrag, false, bf[c4], (short)0, acc[c4], false, false);
      }
    }
    __syncthreads();
    cur ^= 1;
  }

  // Epilogue: D elem i -> row i (lanes 0-15) / 8+i (lanes 16-31), col = lane&15
#pragma unroll
  for (int c4 = 0; c4 < 4; ++c4) {
    int col = colBase + (c4 << 4) + (lane & 15);
    float bval = bias[col];
#pragma unroll
    for (int i = 0; i < 8; ++i) {
      int row = rowBase + (wave << 4) + ((lane & 16) ? 8 + i : i);
      if (row < M) {
        float v = acc[c4][i] + bval;
        if (actC == 1) v = fmaxf(v, 0.0f);
        C[(long)row * Nout + col] = v;
      }
    }
  }
}

// ---------------------------------------------------------------------------
// Per-head relation transform: y[n,h,:] = x[n,h,:] @ W[h] (W: [HEADS,32,32])
// ---------------------------------------------------------------------------
__global__ void rel_transform_kernel(const float* __restrict__ xin,
                                     const float* __restrict__ W,
                                     float* __restrict__ yout, int N)
{
  int gid = blockIdx.x * blockDim.x + threadIdx.x;
  if (gid >= N * HHID) return;
  int n  = gid >> 8;
  int he = gid & 255;
  int hh = he >> 5;
  int e  = he & 31;
  const float* xp = xin + (size_t)n * HHID + hh * DHEAD;
  const float* wp = W + hh * (DHEAD * DHEAD) + e;
  float acc = 0.0f;
#pragma unroll
  for (int d = 0; d < DHEAD; ++d) acc += xp[d] * wp[d * DHEAD];
  yout[gid] = acc;
}

// ---------------------------------------------------------------------------
// Edge pipeline (per relation): logits + segment max, exp + segment sum,
// weighted scatter-add of messages (segment softmax per relation, per head).
// ---------------------------------------------------------------------------
__global__ void edge_logits_kernel(const int* __restrict__ esrc,
                                   const int* __restrict__ edst,
                                   const float* __restrict__ qdt,
                                   const float* __restrict__ kr,
                                   const float* __restrict__ prel,
                                   float* __restrict__ logits,
                                   unsigned* __restrict__ mbuf, int E)
{
  int gid = blockIdx.x * blockDim.x + threadIdx.x;
  if (gid >= E * HEADS) return;
  int e = gid >> 3, hh = gid & 7;
  int s = esrc[e], d = edst[e];
  const float* qp = qdt + (size_t)d * HHID + hh * DHEAD;
  const float* kp = kr  + (size_t)s * HHID + hh * DHEAD;
  float dot = 0.0f;
#pragma unroll
  for (int j = 0; j < DHEAD; ++j) dot += qp[j] * kp[j];
  float lg = dot * prel[hh] * 0.17677669529663687f;  // 1/sqrt(32)
  logits[gid] = lg;
  atomicMax(&mbuf[d * HEADS + hh], fmap(lg));
}

__global__ void edge_exp_kernel(const int* __restrict__ edst,
                                float* __restrict__ logits,
                                const unsigned* __restrict__ mbuf,
                                float* __restrict__ sbuf, int E)
{
  int gid = blockIdx.x * blockDim.x + threadIdx.x;
  if (gid >= E * HEADS) return;
  int e = gid >> 3, hh = gid & 7;
  int d = edst[e];
  float ex = expf(logits[gid] - funmap(mbuf[d * HEADS + hh]));
  logits[gid] = ex;
  atomicAdd(&sbuf[d * HEADS + hh], ex);
}

__global__ void edge_scatter_kernel(const int* __restrict__ esrc,
                                    const int* __restrict__ edst,
                                    const float* __restrict__ logits,
                                    const float* __restrict__ sbuf,
                                    const float* __restrict__ vr,
                                    float* __restrict__ aggdt, int E)
{
  int gid = blockIdx.x * blockDim.x + threadIdx.x;
  if (gid >= E * HEADS) return;
  int e = gid >> 3, hh = gid & 7;
  int s = esrc[e], d = edst[e];
  float alpha = logits[gid] / sbuf[d * HEADS + hh];
  const float* vp = vr + (size_t)s * HHID + hh * DHEAD;
  float* ap = aggdt + (size_t)d * HHID + hh * DHEAD;
#pragma unroll 8
  for (int j = 0; j < DHEAD; ++j) atomicAdd(&ap[j], vp[j] * alpha);
}

// ---------------------------------------------------------------------------
// Gated skip + residual + LayerNorm + ReLU. One wave (32 lanes) per node,
// each lane owns 8 features. h_new = relu(LN(beta*o + (2-beta)*h))
// ---------------------------------------------------------------------------
__global__ __launch_bounds__(256) void gate_ln_kernel(
    const float* __restrict__ o, float* __restrict__ h,
    const float* __restrict__ skip, const float* __restrict__ ln_g,
    const float* __restrict__ ln_b, int N)
{
  int wid  = blockIdx.x * (blockDim.x >> 5) + (threadIdx.x >> 5);
  int lane = threadIdx.x & 31;
  if (wid >= T_NT * N) return;
  int t = wid / N;
  float beta = 1.0f / (1.0f + expf(-skip[t]));

  const float* op = o + (size_t)wid * HHID;
  float*       hp = h + (size_t)wid * HHID;

  float vals[8];
  float sum = 0.0f;
#pragma unroll
  for (int j = 0; j < 8; ++j) {
    int f = lane * 8 + j;
    float x = beta * op[f] + (2.0f - beta) * hp[f];
    vals[j] = x;
    sum += x;
  }
#pragma unroll
  for (int s = 16; s > 0; s >>= 1) sum += __shfl_xor(sum, s, 32);
  float mu = sum * (1.0f / 256.0f);

  float s2 = 0.0f;
#pragma unroll
  for (int j = 0; j < 8; ++j) { float d = vals[j] - mu; s2 += d * d; }
#pragma unroll
  for (int s = 16; s > 0; s >>= 1) s2 += __shfl_xor(s2, s, 32);
  float rinv = rsqrtf(s2 * (1.0f / 256.0f) + 1e-5f);

  const float* g = ln_g + (size_t)t * HHID;
  const float* b = ln_b + (size_t)t * HHID;
#pragma unroll
  for (int j = 0; j < 8; ++j) {
    int f = lane * 8 + j;
    float y = (vals[j] - mu) * rinv * g[f] + b[f];
    hp[f] = fmaxf(y, 0.0f);
  }
}

// ---------------------------------------------------------------------------
extern "C" void kernel_launch(void* const* d_in, const int* in_sizes, int n_in,
                              void* d_out, int out_size, void* d_ws, size_t ws_size,
                              hipStream_t stream)
{
  (void)n_in; (void)out_size; (void)ws_size;

  const float* x     = (const float*)d_in[0];
  const int*   ei    = (const int*)  d_in[1];
  const float* Win   = (const float*)d_in[2];
  const float* b_in  = (const float*)d_in[3];
  const float* Wk    = (const float*)d_in[4];
  const float* bk    = (const float*)d_in[5];
  const float* Wq    = (const float*)d_in[6];
  const float* bq    = (const float*)d_in[7];
  const float* Wv    = (const float*)d_in[8];
  const float* bv    = (const float*)d_in[9];
  const float* Wa    = (const float*)d_in[10];
  const float* ba    = (const float*)d_in[11];
  const float* skip  = (const float*)d_in[12];
  const float* a_rel = (const float*)d_in[13];
  const float* m_rel = (const float*)d_in[14];
  const float* p_rel = (const float*)d_in[15];
  const float* ln_g  = (const float*)d_in[16];
  const float* ln_b  = (const float*)d_in[17];
  const float* Wout  = (const float*)d_in[18];
  const float* bout  = (const float*)d_in[19];
  float* out = (float*)d_out;

  const int N = in_sizes[0] / (T_NT * FIND);
  const int E = in_sizes[1] / (R_ET * 2);

  static const int SRC_T[R_ET] = {0, 1, 1, 1};
  static const int DST_T[R_ET] = {1, 0, 1, 2};

  // ---- workspace carve (fp32 first, then f16 regions; all 16B aligned) ----
  float* ws = (float*)d_ws;
  size_t off = 0;
  const size_t tnh = (size_t)T_NT * N * HHID;
  const size_t nh  = (size_t)N * HHID;
  float* hbuf = ws + off; off += tnh;   // activations [T,N,256] fp32
  float* kbuf = ws + off; off += tnh;   // k; reused as Wa-output after edges
  float* qbuf = ws + off; off += tnh;
  float* vbuf = ws + off; off += tnh;
  float* aggb = ws + off; off += tnh;   // attention aggregation
  float* krb  = ws + off; off += nh;    // relation-transformed k
  float* vrb  = ws + off; off += nh;    // relation-transformed v
  float* logb = ws + off; off += (size_t)E * HEADS;  // logits -> exp weights
  unsigned* mb = (unsigned*)(ws + off); off += (size_t)N * HEADS; // seg max
  float* sb   = ws + off; off += (size_t)N * HEADS;               // seg sum

  _Float16* f16base = (_Float16*)(ws + off);
  size_t hoff = 0;
  const size_t PAD = 128 * (size_t)HHID;  // tile row-overshoot pad (halves)
  _Float16* actf16 = f16base + hoff; hoff += tnh + PAD;                 // shared act buf
  _Float16* winT   = f16base + hoff; hoff += (size_t)T_NT * FIND * HHID;
  _Float16* wkT    = f16base + hoff; hoff += (size_t)L_LAY * T_NT * HHID * HHID;
  _Float16* wqT    = f16base + hoff; hoff += (size_t)L_LAY * T_NT * HHID * HHID;
  _Float16* wvT    = f16base + hoff; hoff += (size_t)L_LAY * T_NT * HHID * HHID;
  _Float16* waT    = f16base + hoff; hoff += (size_t)L_LAY * T_NT * HHID * HHID;
  _Float16* woutT  = f16base + hoff; hoff += (size_t)HHID * OUTD;

  const dim3 blk(256);
  const int rowBlocks = (N + 127) / 128;
  const int edgeGrid  = (E * HEADS + 255) / 256;
  const int relGrid   = (N * HHID + 255) / 256;

  auto cvtGrid = [](long n4) { return (int)((n4 + 255) / 256); };

  // ---- one-time (per launch) weight convert + transpose to f16 [N,K] ----
  {
    long t1 = (long)T_NT * FIND * HHID;
    wtrans_f16_kernel<<<cvtGrid(t1), blk, 0, stream>>>(Win, winT, FIND, HHID, t1);
    long t2 = (long)L_LAY * T_NT * HHID * HHID;
    wtrans_f16_kernel<<<cvtGrid(t2), blk, 0, stream>>>(Wk, wkT, HHID, HHID, t2);
    wtrans_f16_kernel<<<cvtGrid(t2), blk, 0, stream>>>(Wq, wqT, HHID, HHID, t2);
    wtrans_f16_kernel<<<cvtGrid(t2), blk, 0, stream>>>(Wv, wvT, HHID, HHID, t2);
    wtrans_f16_kernel<<<cvtGrid(t2), blk, 0, stream>>>(Wa, waT, HHID, HHID, t2);
    long t3 = (long)HHID * OUTD;
    wtrans_f16_kernel<<<cvtGrid(t3), blk, 0, stream>>>(Wout, woutT, HHID, OUTD, t3);
  }

  // ---- Input projection + ReLU: h = relu(x @ Win + b_in) ----
  {
    long xn4 = (long)T_NT * N * FIND / 4;
    cvt_f16_kernel<<<cvtGrid(xn4), blk, 0, stream>>>(x, actf16, xn4, 0);
    gemm_wmma_kernel<<<dim3(rowBlocks, HHID / 64, T_NT), blk, 0, stream>>>(
        actf16, winT, b_in, hbuf, N, FIND, HHID,
        (long)N * FIND, (long)FIND * HHID, (long)HHID, (long)N * HHID, 1);
  }

  for (int l = 0; l < L_LAY; ++l) {
    const long wOff = (long)l * T_NT * HHID * HHID;
    const long bOff = (long)l * T_NT * HHID;

    // h -> f16 once, consumed by all three projections
    cvt_f16_kernel<<<cvtGrid((long)tnh / 4), blk, 0, stream>>>(
        hbuf, actf16, (long)tnh / 4, 0);

    gemm_wmma_kernel<<<dim3(rowBlocks, HHID / 64, T_NT), blk, 0, stream>>>(
        actf16, wkT + wOff, bk + bOff, kbuf, N, HHID, HHID,
        (long)N * HHID, (long)HHID * HHID, (long)HHID, (long)N * HHID, 0);
    gemm_wmma_kernel<<<dim3(rowBlocks, HHID / 64, T_NT), blk, 0, stream>>>(
        actf16, wqT + wOff, bq + bOff, qbuf, N, HHID, HHID,
        (long)N * HHID, (long)HHID * HHID, (long)HHID, (long)N * HHID, 0);
    gemm_wmma_kernel<<<dim3(rowBlocks, HHID / 64, T_NT), blk, 0, stream>>>(
        actf16, wvT + wOff, bv + bOff, vbuf, N, HHID, HHID,
        (long)N * HHID, (long)HHID * HHID, (long)HHID, (long)N * HHID, 0);

    hipMemsetAsync(aggb, 0, tnh * sizeof(float), stream);

    for (int r = 0; r < R_ET; ++r) {
      const int st = SRC_T[r], dt = DST_T[r];
      const float* arW = a_rel + (size_t)(l * R_ET + r) * HEADS * DHEAD * DHEAD;
      const float* mrW = m_rel + (size_t)(l * R_ET + r) * HEADS * DHEAD * DHEAD;
      const float* pr  = p_rel + (size_t)(l * R_ET + r) * HEADS;
      const int* esrc = ei + ((size_t)r * 2 + 0) * E;
      const int* edst = ei + ((size_t)r * 2 + 1) * E;

      rel_transform_kernel<<<relGrid, blk, 0, stream>>>(
          kbuf + (size_t)st * N * HHID, arW, krb, N);
      rel_transform_kernel<<<relGrid, blk, 0, stream>>>(
          vbuf + (size_t)st * N * HHID, mrW, vrb, N);

      hipMemsetAsync(mb, 0, (size_t)N * HEADS * sizeof(unsigned), stream);
      hipMemsetAsync(sb, 0, (size_t)N * HEADS * sizeof(float), stream);

      edge_logits_kernel<<<edgeGrid, blk, 0, stream>>>(
          esrc, edst, qbuf + (size_t)dt * N * HHID, krb, pr, logb, mb, E);
      edge_exp_kernel<<<edgeGrid, blk, 0, stream>>>(edst, logb, mb, sb, E);
      edge_scatter_kernel<<<edgeGrid, blk, 0, stream>>>(
          esrc, edst, logb, sb, vrb, aggb + (size_t)dt * N * HHID, E);
    }

    // gelu(agg) -> f16 (fused), then o = geluA @ Wa + ba (output reuses kbuf)
    cvt_f16_kernel<<<cvtGrid((long)tnh / 4), blk, 0, stream>>>(
        aggb, actf16, (long)tnh / 4, 1);
    gemm_wmma_kernel<<<dim3(rowBlocks, HHID / 64, T_NT), blk, 0, stream>>>(
        actf16, waT + wOff, ba + bOff, kbuf, N, HHID, HHID,
        (long)N * HHID, (long)HHID * HHID, (long)HHID, (long)N * HHID, 0);

    // gated skip + residual + LN + ReLU (in-place on h)
    gate_ln_kernel<<<(T_NT * N + 7) / 8, blk, 0, stream>>>(
        kbuf, hbuf, skip + (size_t)l * T_NT,
        ln_g + (size_t)l * T_NT * HHID, ln_b + (size_t)l * T_NT * HHID, N);
  }

  // ---- Output projection: out = h @ Wout + bout (weights shared over types) ----
  cvt_f16_kernel<<<cvtGrid((long)tnh / 4), blk, 0, stream>>>(
      hbuf, actf16, (long)tnh / 4, 0);
  gemm_wmma_kernel<<<dim3(rowBlocks, OUTD / 64, T_NT), blk, 0, stream>>>(
      actf16, woutT, bout, out, N, HHID, OUTD,
      (long)N * HHID, 0L, 0L, (long)N * OUTD, 0);
}
